// GatedAttentionUnit_18769007084652
// MI455X (gfx1250) — matrix operand order repaired and checked by reading, not compiled, not run on hardware
//
#include <hip/hip_runtime.h>
#include <hip/hip_bf16.h>

// Problem constants
#define BB   4
#define SS   4096
#define HH   768
#define UU   1536
#define KEYD 128
#define DI   (2*UU + KEYD)   // 3200
#define MM   (BB*SS)         // 16384

typedef __attribute__((ext_vector_type(16))) __bf16 v16bf;
typedef __attribute__((ext_vector_type(8)))  float  v8f;
typedef __attribute__((ext_vector_type(4)))  unsigned v4u;
typedef __attribute__((ext_vector_type(8)))  int      v8i;
typedef __attribute__((ext_vector_type(4)))  int      v4i;

#if defined(__AMDGCN__) && __has_builtin(__builtin_amdgcn_tensor_load_to_lds)
#define USE_TDM 1
#else
#define USE_TDM 0
#endif

__device__ __forceinline__ v8f wmma_bf16(v16bf a, v16bf b, v8f c) {
  return __builtin_amdgcn_wmma_f32_16x16x32_bf16(
      false, a, false, b, (short)0, c, false, false);
}

// bf16 (packed pair in a dword) -> f32, exact
__device__ __forceinline__ float bflo(unsigned d) { return __uint_as_float(d << 16); }
__device__ __forceinline__ float bfhi(unsigned d) { return __uint_as_float(d & 0xFFFF0000u); }

// Fill 8 fragment elements [e0..e0+7] from one uint4 of packed bf16 with affine g*x+b
__device__ __forceinline__ void unpack_affine(uint4 d, const float* __restrict__ g,
                                              const float* __restrict__ b, int K0,
                                              v16bf& f, int e0) {
  unsigned dw[4] = {d.x, d.y, d.z, d.w};
#pragma unroll
  for (int w = 0; w < 4; ++w) {
    const int K = K0 + 2*w;
    f[e0 + 2*w]     = (__bf16)(bflo(dw[w]) * g[K]     + b[K]);
    f[e0 + 2*w + 1] = (__bf16)(bfhi(dw[w]) * g[K + 1] + b[K + 1]);
  }
}

#if USE_TDM
// Tensor Data Mover: 2D tile (tile0 x tile1 elements, row stride stride0 elements)
// from global -> LDS at byte offset lds_byte.  dsz_code: 0=1B,1=2B,2=4B.
// Descriptor packing per CDNA5 ISA.
// This toolchain's builtin arity is 6: (g0, g1, g2, g3, g_extra8, cpol).
__device__ __forceinline__ void tdm_load_2d(const void* gptr, unsigned lds_byte,
                                            unsigned tile0, unsigned tile1,
                                            unsigned long long stride0,
                                            unsigned dsz_code) {
  const unsigned long long ga = (unsigned long long)(uintptr_t)gptr;
  v4u g0;
  g0.x = 1u;                                                 // count=1, user mode
  g0.y = lds_byte;                                           // lds_addr[31:0]
  g0.z = (unsigned)(ga & 0xFFFFFFFFu);                       // global_addr[31:0]
  g0.w = (unsigned)((ga >> 32) & 0x1FFFFFFu) | (2u << 30);   // addr[56:32] | type=2
  const unsigned td0 = tile0, td1 = tile1;                   // tensor dims == tile dims
  v8i g1;
  g1[0] = (int)(dsz_code << 16);                             // wg_mask=0, data_size
  g1[1] = (int)((td0 & 0xFFFFu) << 16);                      // tensor_dim0[15:0] @48
  g1[2] = (int)((td0 >> 16) | ((td1 & 0xFFFFu) << 16));      // td0[31:16], td1[15:0]
  g1[3] = (int)((td1 >> 16) | (tile0 << 16));                // td1[31:16], tile_dim0
  g1[4] = (int)(tile1 & 0xFFFFu);                            // tile_dim1, tile_dim2=0
  g1[5] = (int)(stride0 & 0xFFFFFFFFull);                    // dim0_stride[31:0]
  g1[6] = (int)((stride0 >> 32) & 0xFFFFull);                // dim0_stride[47:32]
  g1[7] = 0;
  const v4i z4 = {0, 0, 0, 0};
  const v8i z8 = {0, 0, 0, 0, 0, 0, 0, 0};
  __builtin_amdgcn_tensor_load_to_lds(g0, g1, z4, z4, z8, 0);
}
#endif

// ---------------------------------------------------------------------------
// Kernel 1: h = silu(x @ Wi + bi), bf16 out.  One wave = one 16x16 tile;
// B (Wi) panel staged per wave via TDM (strided fp32 rows -> LDS).
// ---------------------------------------------------------------------------
__global__ __launch_bounds__(256)
void k_in_gemm(const float* __restrict__ x, const float* __restrict__ Wi,
               const float* __restrict__ bi, __bf16* __restrict__ h) {
#if USE_TDM
  __shared__ __align__(16) float sB[8][32][16];   // per-wave 2KB B panel
#endif
  const int lane = threadIdx.x & 31;
  const int wv8  = threadIdx.x >> 5;
  const int wv   = blockIdx.x * 8 + wv8;
  const int tM   = wv / (DI/16);
  const int tN   = wv % (DI/16);
  const int half = lane >> 4, r = lane & 15;
  const int rowA = tM*16 + r;
  const int colB = tN*16 + r;

  v8f acc = {};
  for (int kb = 0; kb < HH; kb += 32) {
    v16bf af, bf;
#pragma unroll
    for (int j = 0; j < 8; ++j) {           // A: x fp32, float2 pairs
      const int k0 = kb + half*16 + j*2;
      const float2 p = *(const float2*)(x + (size_t)rowA*HH + k0);
      af[2*j]   = (__bf16)p.x;
      af[2*j+1] = (__bf16)p.y;
    }
#if USE_TDM
    asm volatile("s_wait_dscnt 0" ::: "memory");       // prior DS reads done
    tdm_load_2d(Wi + (size_t)kb*DI + tN*16,
                (unsigned)(uintptr_t)&sB[wv8][0][0], 16, 32, DI, 2);
    __builtin_amdgcn_s_wait_tensorcnt(0);
#pragma unroll
    for (int e = 0; e < 16; ++e)
      bf[e] = (__bf16)sB[wv8][half*16 + e][r];
#else
#pragma unroll
    for (int e = 0; e < 16; ++e)
      bf[e] = (__bf16)Wi[(size_t)(kb + half*16 + e)*DI + colB];
#endif
    acc = wmma_bf16(af, bf, acc);
  }
#pragma unroll
  for (int i = 0; i < 8; ++i) {
    const int row = tM*16 + half*8 + i;
    const float s = acc[i] + bi[colB];
    const float o = s * __builtin_amdgcn_rcpf(1.0f + __expf(-s));  // fast silu
    h[(size_t)row*DI + colB] = (__bf16)o;
  }
}

// ---------------------------------------------------------------------------
// Kernel 2: z = (relu((q k^T)/sqrt(KEY))^2 / S) @ v, fused uz = u*z.
// 8 waves/block; wave owns 192 UNITS columns; v panels staged via TDM.
// ---------------------------------------------------------------------------
__global__ __launch_bounds__(256)
void k_attn(const __bf16* __restrict__ h,
            const float* __restrict__ gq, const float* __restrict__ bq,
            const float* __restrict__ gk, const float* __restrict__ bk,
            __bf16* __restrict__ uz) {
#if USE_TDM
  __shared__ __align__(16) __bf16 sV[8][32][96];  // per-wave 6KB v panel
#endif
  __shared__ __align__(16) __bf16 sA[8][16][32];  // per-wave score tile
  const int lane  = threadIdx.x & 31;
  const int wv    = threadIdx.x >> 5;
  const int half  = lane >> 4, r = lane & 15;
  const int batch = blockIdx.x / (SS/16);
  const int mt    = blockIdx.x % (SS/16);
  const size_t rowBase = (size_t)batch * SS;

  // q fragments over KEY=128: vectorized bf16 loads + fp32 affine
  v16bf qf[4];
  {
    const __bf16* qrow = h + (rowBase + (size_t)mt*16 + r)*DI + 2*UU;
#pragma unroll
    for (int ks = 0; ks < 4; ++ks) {
      const int K0 = ks*32 + half*16;
      unpack_affine(*(const uint4*)(qrow + K0),      gq, bq, K0,      qf[ks], 0);
      unpack_affine(*(const uint4*)(qrow + K0 + 8), gq, bq, K0 + 8, qf[ks], 8);
    }
  }

  v8f acc[12];
#pragma unroll
  for (int t = 0; t < 12; ++t) { v8f z = {}; acc[t] = z; }

  const float sc   = 0.08838834764831845f;  // 1/sqrt(128)
  const float invn = 1.0f / (float)SS;

  for (int nb = 0; nb < SS; nb += 32) {
    // ---- scores for 32 keys ----
    v8f s0 = {}, s1 = {};
    const __bf16* krow0 = h + (rowBase + nb + r)*DI + 2*UU;
    const __bf16* krow1 = krow0 + (size_t)16*DI;
#pragma unroll
    for (int ks = 0; ks < 4; ++ks) {
      v16bf kf0, kf1;
      const int K0 = ks*32 + half*16;       // B-frag: 16 contiguous K
      unpack_affine(*(const uint4*)(krow0 + K0),     gk, bk, K0,     kf0, 0);
      unpack_affine(*(const uint4*)(krow0 + K0 + 8), gk, bk, K0 + 8, kf0, 8);
      unpack_affine(*(const uint4*)(krow1 + K0),     gk, bk, K0,     kf1, 0);
      unpack_affine(*(const uint4*)(krow1 + K0 + 8), gk, bk, K0 + 8, kf1, 8);
      s0 = wmma_bf16(qf[ks], kf0, s0);
      s1 = wmma_bf16(qf[ks], kf1, s1);
    }
    // ---- relu^2/n in fp32; stage as bf16 A-fragment source in LDS ----
#pragma unroll
    for (int i = 0; i < 8; ++i) {
      float a0 = fmaxf(s0[i]*sc, 0.0f); a0 = a0*a0*invn;
      float a1 = fmaxf(s1[i]*sc, 0.0f); a1 = a1*a1*invn;
      sA[wv][half*8 + i][r]      = (__bf16)a0;
      sA[wv][half*8 + i][16 + r] = (__bf16)a1;
    }
    asm volatile("s_wait_dscnt 0" ::: "memory");
    v16bf af;
#pragma unroll
    for (int e = 0; e < 16; ++e) {
      const int j = e >> 1, p = e & 1;
      const int K = half*16 + 2*j + p;
      af[e] = sA[wv][r][K];
    }
    // ---- z += A @ v over this wave's 192 columns (2 TDM panels of 96) ----
#pragma unroll
    for (int g = 0; g < 2; ++g) {
#if USE_TDM
      asm volatile("s_wait_dscnt 0" ::: "memory");  // panel reads done
      tdm_load_2d(h + (rowBase + nb)*DI + UU + wv*192 + g*96,
                  (unsigned)(uintptr_t)&sV[wv][0][0], 96, 32, DI, 1);
      __builtin_amdgcn_s_wait_tensorcnt(0);
#endif
#pragma unroll
      for (int t = 0; t < 6; ++t) {
        v16bf vf;
#if USE_TDM
#pragma unroll
        for (int e = 0; e < 16; ++e)
          vf[e] = sV[wv][half*16 + e][t*16 + r];
#else
        const int ucol = wv*192 + g*96 + t*16 + r;
#pragma unroll
        for (int e = 0; e < 16; ++e)
          vf[e] = h[(rowBase + nb + half*16 + e)*DI + UU + ucol];
#endif
        acc[g*6 + t] = wmma_bf16(af, vf, acc[g*6 + t]);
      }
    }
  }

  // epilogue: uz = u * z
#pragma unroll
  for (int g = 0; g < 2; ++g)
#pragma unroll
    for (int t = 0; t < 6; ++t) {
      const int ucol = wv*192 + g*96 + t*16 + r;
#pragma unroll
      for (int i = 0; i < 8; ++i) {
        const size_t row = rowBase + (size_t)mt*16 + half*8 + i;
        const float u = (float)h[row*DI + ucol];
        uz[row*UU + ucol] = (__bf16)(u * acc[g*6 + t][i]);
      }
    }
}

// ---------------------------------------------------------------------------
// Kernel 3: o = (u*z) @ Wo + bo, fp32 output.  Wo panel staged via TDM.
// ---------------------------------------------------------------------------
__global__ __launch_bounds__(256)
void k_out_gemm(const __bf16* __restrict__ uz, const float* __restrict__ Wo,
                const float* __restrict__ bo, float* __restrict__ out) {
#if USE_TDM
  __shared__ __align__(16) float sB[8][32][16];
#endif
  const int lane = threadIdx.x & 31;
  const int wv8  = threadIdx.x >> 5;
  const int wv   = blockIdx.x * 8 + wv8;
  const int tM   = wv / (HH/16);
  const int tN   = wv % (HH/16);
  const int half = lane >> 4, r = lane & 15;
  const int rowA = tM*16 + r;
  const int colB = tN*16 + r;

  v8f acc = {};
  for (int kb = 0; kb < UU; kb += 32) {
    // A: uz bf16 rows — two aligned 16B runs fill the fragment exactly
    union { v16bf v; uint4 q[2]; } A;
    A.q[0] = *(const uint4*)(uz + (size_t)rowA*UU + kb + half*16);
    A.q[1] = *(const uint4*)(uz + (size_t)rowA*UU + kb + half*16 + 8);
    v16bf bf;
#if USE_TDM
    asm volatile("s_wait_dscnt 0" ::: "memory");
    tdm_load_2d(Wo + (size_t)kb*HH + tN*16,
                (unsigned)(uintptr_t)&sB[wv8][0][0], 16, 32, HH, 2);
    __builtin_amdgcn_s_wait_tensorcnt(0);
#pragma unroll
    for (int e = 0; e < 16; ++e)
      bf[e] = (__bf16)sB[wv8][half*16 + e][r];
#else
#pragma unroll
    for (int e = 0; e < 16; ++e)
      bf[e] = (__bf16)Wo[(size_t)(kb + half*16 + e)*HH + colB];
#endif
    acc = wmma_bf16(A.v, bf, acc);
  }
#pragma unroll
  for (int i = 0; i < 8; ++i) {
    const int row = tM*16 + half*8 + i;
    out[(size_t)row*HH + colB] = acc[i] + bo[colB];
  }
}

// ---------------------------------------------------------------------------
extern "C" void kernel_launch(void* const* d_in, const int* in_sizes, int n_in,
                              void* d_out, int out_size, void* d_ws, size_t ws_size,
                              hipStream_t stream) {
  const float* x  = (const float*)d_in[0];
  const float* Wi = (const float*)d_in[1];
  const float* bi = (const float*)d_in[2];
  const float* gq = (const float*)d_in[3];
  const float* bq = (const float*)d_in[4];
  const float* gk = (const float*)d_in[5];
  const float* bk = (const float*)d_in[6];
  const float* Wo = (const float*)d_in[7];
  const float* bo = (const float*)d_in[8];
  float* out = (float*)d_out;

  __bf16* h  = (__bf16*)d_ws;               // MM*DI bf16  (100 MB)
  __bf16* uz = h + (size_t)MM * DI;         // MM*UU bf16  ( 48 MB)

  k_in_gemm<<<(MM/16)*(DI/16)/8, 256, 0, stream>>>(x, Wi, bi, h);
  k_attn<<<BB*(SS/16), 256, 0, stream>>>(h, gq, bq, gk, bk, uz);
  k_out_gemm<<<(MM/16)*(HH/16)/8, 256, 0, stream>>>(uz, Wo, bo, out);
}
